// EculideanAttention_73899207295099
// MI455X (gfx1250) — compile-verified, matrix operand order checked by reading, not run on hardware
//
#include <hip/hip_runtime.h>
#include <math.h>

#define N_NODES   16384
#define N_EDGES   262144
#define NODE_DIM  120
#define EQUI_DIM  240
#define NUM_IRREPS 112
#define NUM_BASIS 20
#define KP        128                      // padded inner dim for WMMA
#define ASTR      136                      // LDS row stride (bank-rotating, 16B aligned)
#define TE        32                       // edges per block
#define EPSV      1e-5f
#define SCALE_SCALAR 0.09128709291752768f  // 1/sqrt(120)
#define SCALE_EQUI   0.09449111825230679f  // 1/sqrt(112)

typedef __attribute__((ext_vector_type(16))) __bf16 v16bf;
typedef __attribute__((ext_vector_type(8)))  __bf16 v8bf;
typedef __attribute__((ext_vector_type(8)))  float  v8f;

static __device__ __forceinline__ __bf16 f2bf(float x) {
  unsigned u = __builtin_bit_cast(unsigned, x);
  unsigned r = u + 0x7FFFu + ((u >> 16) & 1u);   // round-to-nearest-even
  unsigned short h = (unsigned short)(r >> 16);
  return __builtin_bit_cast(__bf16, h);
}
static __device__ __forceinline__ float bf2f(__bf16 b) {
  unsigned short h = __builtin_bit_cast(unsigned short, b);
  unsigned u = ((unsigned)h) << 16;
  return __builtin_bit_cast(float, u);
}

// Build a 16x32 A (or 32x16 B) bf16 fragment for lane (hf, m/n) from a
// row-major run of 32 bf16 values: two contiguous 16-byte chunks.
static __device__ __forceinline__ v16bf frag16(const __bf16* row, int k0, int hf) {
  v8bf lo = *(const v8bf*)(row + k0 + hf * 8);
  v8bf hi = *(const v8bf*)(row + k0 + 16 + hf * 8);
  v16bf a;
#pragma unroll
  for (int i = 0; i < 8; ++i) { a[i] = lo[i]; a[i + 8] = hi[i]; }
  return a;
}

static __device__ __forceinline__ v8f wmma_bf16(v16bf a, v16bf b, v8f c) {
  // (neg_a, A, neg_b, B, c_mod, C, reuse_a, reuse_b)
  return __builtin_amdgcn_wmma_f32_16x16x32_bf16(false, a, false, b, (short)0, c, false, false);
}

// ---------------------------------------------------------------------------
// Pack fp32 weight W[Kin][Nout] -> bf16 WMMA-B layout dst[n][k], n<128, k<Kt,
// zero padded. Each B-fragment load becomes two contiguous b128 loads.
// ---------------------------------------------------------------------------
__global__ void pack_weight(const float* __restrict__ src, __bf16* __restrict__ dst,
                            int Kin, int Nout, int Kt) {
  int i = blockIdx.x * blockDim.x + threadIdx.x;
  if (i >= 128 * Kt) return;
  int n = i / Kt, k = i - n * Kt;
  float v = (k < Kin && n < Nout) ? src[k * Nout + n] : 0.0f;
  dst[i] = f2bf(v);
}

// ---------------------------------------------------------------------------
// Per-node: LayerNorm -> bf16 ns_pack[N][128]; O3-LayerNorm -> bf16 ne[N][240];
// initialize d_out with the residual (node_scalar | node_equi).
// ---------------------------------------------------------------------------
__global__ void node_prep(const float* __restrict__ xs, const float* __restrict__ xe,
                          const float* __restrict__ lnw, const float* __restrict__ lnb,
                          const float* __restrict__ o3w,
                          __bf16* __restrict__ ns_pack, __bf16* __restrict__ ne,
                          float* __restrict__ out) {
  int n = blockIdx.x * blockDim.x + threadIdx.x;
  if (n >= N_NODES) return;

  const float* x = xs + (size_t)n * NODE_DIM;
  float s = 0.f, s2 = 0.f;
  for (int j = 0; j < NODE_DIM; ++j) { float v = x[j]; s += v; s2 += v * v; }
  float mu  = s * (1.f / NODE_DIM);
  float var = s2 * (1.f / NODE_DIM) - mu * mu;
  float inv = rsqrtf(var + EPSV);
  __bf16* np = ns_pack + (size_t)n * KP;
  for (int j = 0; j < NODE_DIM; ++j) np[j] = f2bf((x[j] - mu) * inv * lnw[j] + lnb[j]);
  for (int j = NODE_DIM; j < KP; ++j) np[j] = f2bf(0.f);

  // residual init of outputs
  float* os = out + (size_t)n * NODE_DIM;
  for (int j = 0; j < NODE_DIM; ++j) os[j] = x[j];
  const float* e = xe + (size_t)n * EQUI_DIM;
  float* oe = out + (size_t)N_NODES * NODE_DIM + (size_t)n * EQUI_DIM;
  for (int j = 0; j < EQUI_DIM; ++j) oe[j] = e[j];

  // O3 layer norm -> bf16
  __bf16* eo = ne + (size_t)n * EQUI_DIM;
  // l=0: comps [0,64), mean-center over mul
  float m0 = 0.f;
  for (int i = 0; i < 64; ++i) m0 += e[i];
  m0 *= (1.f / 64.f);
  float ss = 0.f;
  for (int i = 0; i < 64; ++i) { float t = e[i] - m0; ss += t * t; }
  float r0 = rsqrtf(ss * (1.f / 64.f) + EPSV);
  for (int i = 0; i < 64; ++i) eo[i] = f2bf((e[i] - m0) * r0 * o3w[i]);
  // l=1: comps [64,160), mul 32, d 3
  ss = 0.f;
  for (int c = 64; c < 160; ++c) ss += e[c] * e[c];
  float r1 = rsqrtf(ss * (1.f / 96.f) + EPSV);
  for (int m = 0; m < 32; ++m) {
    float w = r1 * o3w[64 + m];
    for (int d = 0; d < 3; ++d) { int c = 64 + m * 3 + d; eo[c] = f2bf(e[c] * w); }
  }
  // l=2: comps [160,240), mul 16, d 5
  ss = 0.f;
  for (int c = 160; c < 240; ++c) ss += e[c] * e[c];
  float r2 = rsqrtf(ss * (1.f / 80.f) + EPSV);
  for (int m = 0; m < 16; ++m) {
    float w = r2 * o3w[96 + m];
    for (int d = 0; d < 5; ++d) { int c = 160 + m * 5 + d; eo[c] = f2bf(e[c] * w); }
  }
}

// ---------------------------------------------------------------------------
// Node projections: proj[m][N][128] (bf16) = ns @ W_m  (m = q,k,v,qs,ks,vs).
// One block = one 16-node row tile x one matrix; one wave per 16-col tile.
// ---------------------------------------------------------------------------
__global__ void __launch_bounds__(256) proj_gemm(const __bf16* __restrict__ ns_pack,
                                                 const __bf16* __restrict__ pw,
                                                 __bf16* __restrict__ proj) {
  int bid = blockIdx.x;
  int rt = bid / 6, m = bid - rt * 6;
  int lane = threadIdx.x & 31, wv = threadIdx.x >> 5;
  int hf = lane >> 4, ln = lane & 15;
  int nt = wv;  // 8 waves -> 8 col tiles (128 cols)

  const __bf16* arow = ns_pack + (size_t)(rt * 16 + ln) * KP;
  const __bf16* brow = pw + (size_t)m * KP * KP + (size_t)(nt * 16 + ln) * KP;
  v8f acc = {};
#pragma unroll
  for (int ksx = 0; ksx < 4; ++ksx) {
    v16bf a = frag16(arow, ksx * 32, hf);
    v16bf b = frag16(brow, ksx * 32, hf);
    acc = wmma_bf16(a, b, acc);
  }
  __bf16* dst = proj + (size_t)m * N_NODES * KP;
  int col = nt * 16 + ln;
#pragma unroll
  for (int r = 0; r < 8; ++r) {
    int row = rt * 16 + hf * 8 + r;     // C/D layout: M = r + 8*half, N = lane&15
    dst[(size_t)row * KP + col] = f2bf(acc[r]);
  }
}

// ---------------------------------------------------------------------------
// Edge kernel: 1 block = 32 edges, 256 threads (8 waves).
// equi_dot -> bf16 LDS A tile; two-layer MLP via WMMA; attention + atomic scatter.
// ---------------------------------------------------------------------------
__global__ void __launch_bounds__(256) edge_kernel(
    const int* __restrict__ eidx, const float* __restrict__ rbf,
    const float* __restrict__ fcut, const float* __restrict__ rsh,
    const __bf16* __restrict__ ne, const __bf16* __restrict__ proj,
    const __bf16* __restrict__ pw_inv1, const __bf16* __restrict__ pw_rbf1,
    const __bf16* __restrict__ pw_inv2, const __bf16* __restrict__ pw_rbf2,
    const float* __restrict__ inv_b1, const float* __restrict__ rbf_b1,
    const float* __restrict__ inv_b2, const float* __restrict__ rbf_b2,
    float* __restrict__ out) {
  __shared__ int   s_ct[TE], s_nb[TE];
  __shared__ float s_fc[TE];
  __shared__ __align__(16) __bf16 s_inv[TE][ASTR];
  __shared__ __align__(16) __bf16 s_rbf[TE][40];
  __shared__ __align__(16) __bf16 s_h1[TE][ASTR];
  __shared__ __align__(16) __bf16 s_h1r[TE][ASTR];
  __shared__ float s_w[TE][NODE_DIM];

  const int tid = threadIdx.x;
  const int ge0 = blockIdx.x * TE;
  if (tid < TE) {
    s_ct[tid] = eidx[ge0 + tid];
    s_nb[tid] = eidx[N_EDGES + ge0 + tid];
    s_fc[tid] = fcut[ge0 + tid];
  }
  // Prefetch the gather rows consumed later (overlap with equi_dot + 2 GEMMs).
  if (tid < 192) {            // 6 projection arrays x 32 edges
    int m = tid >> 5, e = tid & 31;
    int node = (m == 0 || m == 3) ? eidx[ge0 + e] : eidx[N_EDGES + ge0 + e];
    __builtin_prefetch(proj + ((size_t)m * N_NODES + node) * KP, 0, 3);
  } else {                    // 2 ne rows x 32 edges
    int q = tid - 192;
    int e = q & 31;
    int node = (q >> 5) ? eidx[N_EDGES + ge0 + e] : eidx[ge0 + e];
    __builtin_prefetch(ne + (size_t)node * EQUI_DIM, 0, 3);
  }
  __syncthreads();

  // rbf tile (K padded to 32)
  for (int idx = tid; idx < TE * 32; idx += 256) {
    int e = idx >> 5, k = idx & 31;
    s_rbf[e][k] = (k < NUM_BASIS) ? f2bf(rbf[(size_t)(ge0 + e) * NUM_BASIS + k]) : f2bf(0.f);
  }
  // equi_dot invariants of x_ij = ne[nb] - ne[ct]  (K padded to 128)
  for (int idx = tid; idx < TE * 128; idx += 256) {
    int e = idx >> 7, w = idx & 127;
    float sv = 0.f;
    if (w < NUM_IRREPS) {
      const __bf16* en = ne + (size_t)s_nb[e] * EQUI_DIM;
      const __bf16* ec = ne + (size_t)s_ct[e] * EQUI_DIM;
      int base, d;
      if (w < 64)      { base = w;                  d = 1; }
      else if (w < 96) { base = 64 + (w - 64) * 3;  d = 3; }
      else             { base = 160 + (w - 96) * 5; d = 5; }
      for (int c = 0; c < d; ++c) { float x = bf2f(en[base + c]) - bf2f(ec[base + c]); sv += x * x; }
    }
    s_inv[e][w] = f2bf(sv);
  }
  __syncthreads();

  const int lane = tid & 31, wv = tid >> 5;
  const int hf = lane >> 4, ln = lane & 15;

  // ---- layer 1: h1 = silu(inv@inv_w1 + b1), h1r = silu(rbf@rbf_w1 + b1r) ----
  // 32 tasks: [br(2)][rt(2)][nt(8)], 4 per wave
#pragma unroll
  for (int t = 0; t < 4; ++t) {
    int task = wv * 4 + t;
    int br = task >> 4, rem = task & 15;
    int rt = rem >> 3, nt = rem & 7;
    v8f acc = {};
    if (br == 0) {
      const __bf16* brow = pw_inv1 + (size_t)(nt * 16 + ln) * KP;
#pragma unroll
      for (int ksx = 0; ksx < 4; ++ksx) {
        v16bf a = frag16(&s_inv[rt * 16 + ln][0], ksx * 32, hf);
        v16bf b = frag16(brow, ksx * 32, hf);
        acc = wmma_bf16(a, b, acc);
      }
    } else {
      const __bf16* brow = pw_rbf1 + (size_t)(nt * 16 + ln) * 32;
      v16bf a = frag16(&s_rbf[rt * 16 + ln][0], 0, hf);
      v16bf b = frag16(brow, 0, hf);
      acc = wmma_bf16(a, b, acc);
    }
    const float* bias = br ? rbf_b1 : inv_b1;
    __bf16* dst = br ? &s_h1r[0][0] : &s_h1[0][0];
    int col = nt * 16 + ln;
#pragma unroll
    for (int r = 0; r < 8; ++r) {
      int m = rt * 16 + hf * 8 + r;  // edge row
      float v = 0.f;
      if (col < NODE_DIM) {
        float xx = acc[r] + bias[col];
        v = xx / (1.f + __expf(-xx));  // SiLU
      }
      dst[m * ASTR + col] = f2bf(v);
    }
  }
  __syncthreads();

  // ---- layer 2: w_ij = (h1@inv_w2 + h1r@rbf_w2 + b2a + b2b) * fcut ----
  // 16 tasks: [rt(2)][nt(8)], 2 per wave
#pragma unroll
  for (int t = 0; t < 2; ++t) {
    int task = wv * 2 + t;
    int rt = task >> 3, nt = task & 7;
    v8f acc = {};
    const __bf16* b1r = pw_inv2 + (size_t)(nt * 16 + ln) * KP;
    const __bf16* b2r = pw_rbf2 + (size_t)(nt * 16 + ln) * KP;
#pragma unroll
    for (int ksx = 0; ksx < 4; ++ksx) {
      v16bf a = frag16(&s_h1[rt * 16 + ln][0], ksx * 32, hf);
      v16bf b = frag16(b1r, ksx * 32, hf);
      acc = wmma_bf16(a, b, acc);
    }
#pragma unroll
    for (int ksx = 0; ksx < 4; ++ksx) {
      v16bf a = frag16(&s_h1r[rt * 16 + ln][0], ksx * 32, hf);
      v16bf b = frag16(b2r, ksx * 32, hf);
      acc = wmma_bf16(a, b, acc);
    }
    int col = nt * 16 + ln;
    if (col < NODE_DIM) {
      float bsum = inv_b2[col] + rbf_b2[col];
#pragma unroll
      for (int r = 0; r < 8; ++r) {
        int m = rt * 16 + hf * 8 + r;
        s_w[m][col] = (acc[r] + bsum) * s_fc[m];
      }
    }
  }
  __syncthreads();

  // ---- attention + scatter: 8 lanes per edge, interleaved j = r + 8*jj ----
  {
    int e = tid >> 3, r = tid & 7;
    int ge = ge0 + e;
    int ct = s_ct[e], nb = s_nb[e];
    float fc = s_fc[e];
    const __bf16* qi  = proj + (size_t)(0 * N_NODES + ct) * KP;
    const __bf16* ki  = proj + (size_t)(1 * N_NODES + nb) * KP;
    const __bf16* vi  = proj + (size_t)(2 * N_NODES + nb) * KP;
    const __bf16* qsp = proj + (size_t)(3 * N_NODES + ct) * KP;
    const __bf16* kep = proj + (size_t)(4 * N_NODES + nb) * KP;
    const __bf16* vsp = proj + (size_t)(5 * N_NODES + nb) * KP;
    float* out_s = out;
    float* out_e = out + (size_t)N_NODES * NODE_DIM;

    // scalar multi-head attention: 4 per-head accumulators, full 8-lane reduce
    float h0 = 0.f, h1 = 0.f, h2 = 0.f, h3 = 0.f;
    float a0 = 0.f, a1 = 0.f, a2 = 0.f;   // equivariant per-l accumulators
#pragma unroll
    for (int jj = 0; jj < 15; ++jj) {
      int j = r + 8 * jj;
      float wj = s_w[e][j];
      float ts = bf2f(qi[j]) * wj * bf2f(ki[j]);
      if (j < 30) h0 += ts; else if (j < 60) h1 += ts; else if (j < 90) h2 += ts; else h3 += ts;
      float te = bf2f(qsp[j]) * wj * bf2f(kep[j]);
      if (j < 40) a0 += te; else if (j < 80) a1 += te; else a2 += te;
    }
#pragma unroll
    for (int mk = 1; mk < 8; mk <<= 1) {
      h0 += __shfl_xor(h0, mk); h1 += __shfl_xor(h1, mk);
      h2 += __shfl_xor(h2, mk); h3 += __shfl_xor(h3, mk);
      a0 += __shfl_xor(a0, mk); a1 += __shfl_xor(a1, mk);
      a2 += __shfl_xor(a2, mk);
    }
    h0 *= SCALE_SCALAR; h1 *= SCALE_SCALAR; h2 *= SCALE_SCALAR; h3 *= SCALE_SCALAR;
    a0 *= SCALE_EQUI;   a1 *= SCALE_EQUI;   a2 *= SCALE_EQUI;

#pragma unroll
    for (int jj = 0; jj < 15; ++jj) {
      int j = r + 8 * jj;
      float attn = (j < 30) ? h0 : (j < 60) ? h1 : (j < 90) ? h2 : h3;
      atomicAdd(out_s + (size_t)ct * NODE_DIM + j, attn * bf2f(vi[j]));
    }
#pragma unroll
    for (int cc = 0; cc < 30; ++cc) {
      int c = r + 8 * cc;  // covers 0..239
      int w; float al;
      if (c < 64)       { w = c;                   al = a0; }
      else if (c < 160) { w = 64 + (c - 64) / 3;   al = a1; }
      else              { w = 96 + (c - 160) / 5;  al = a2; }
      float val = rsh[(size_t)ge * EQUI_DIM + c] * (al * bf2f(vsp[w])) * fc;
      atomicAdd(out_e + (size_t)ct * EQUI_DIM + c, val);
    }
  }
}

// ---------------------------------------------------------------------------
extern "C" void kernel_launch(void* const* d_in, const int* in_sizes, int n_in,
                              void* d_out, int out_size, void* d_ws, size_t ws_size,
                              hipStream_t stream) {
  const float* node_scalar = (const float*)d_in[0];
  const float* node_equi   = (const float*)d_in[1];
  const float* rbf    = (const float*)d_in[2];
  const float* fcut   = (const float*)d_in[3];
  const float* rsh    = (const float*)d_in[4];
  const float* ln_w   = (const float*)d_in[5];
  const float* ln_b   = (const float*)d_in[6];
  const float* o3w    = (const float*)d_in[7];
  const float* Wq     = (const float*)d_in[8];
  const float* Wk     = (const float*)d_in[9];
  const float* Wv     = (const float*)d_in[10];
  const float* Wqs    = (const float*)d_in[11];
  const float* Wks    = (const float*)d_in[12];
  const float* Wvs    = (const float*)d_in[13];
  const float* rbf_w1 = (const float*)d_in[14];
  const float* rbf_b1 = (const float*)d_in[15];
  const float* rbf_w2 = (const float*)d_in[16];
  const float* rbf_b2 = (const float*)d_in[17];
  const float* inv_w1 = (const float*)d_in[18];
  const float* inv_b1 = (const float*)d_in[19];
  const float* inv_w2 = (const float*)d_in[20];
  const float* inv_b2 = (const float*)d_in[21];
  const int*   eidx   = (const int*)d_in[22];
  (void)in_sizes; (void)n_in; (void)out_size; (void)ws_size;

  // workspace carve (bytes): packed weights | ns_pack bf16 | ne bf16 | proj bf16
  char* ws = (char*)d_ws;
  __bf16* pw_proj = (__bf16*)ws;             // 6 * 128*128 bf16
  __bf16* pw_inv1 = pw_proj + 98304;         // 128*128
  __bf16* pw_rbf1 = pw_proj + 114688;        // 128*32
  __bf16* pw_inv2 = pw_proj + 118784;        // 128*128
  __bf16* pw_rbf2 = pw_proj + 135168;        // 128*128  (end = 151552 elems)
  __bf16* ns_pack = (__bf16*)(ws + 303104);          // N*128 bf16 = 4 MB
  __bf16* nebuf   = (__bf16*)(ws + 4497408);         // N*240 bf16 = 7.9 MB
  __bf16* proj    = (__bf16*)(ws + 12361728);        // 6*N*128 bf16 = 25.2 MB

  // pack weights into bf16 WMMA-B layout, zero padded
  pack_weight<<<64, 256, 0, stream>>>(Wq,     pw_proj + 0 * 16384, 120, 120, 128);
  pack_weight<<<64, 256, 0, stream>>>(Wk,     pw_proj + 1 * 16384, 120, 120, 128);
  pack_weight<<<64, 256, 0, stream>>>(Wv,     pw_proj + 2 * 16384, 120, 120, 128);
  pack_weight<<<64, 256, 0, stream>>>(Wqs,    pw_proj + 3 * 16384, 120, 120, 128);
  pack_weight<<<64, 256, 0, stream>>>(Wks,    pw_proj + 4 * 16384, 120, 120, 128);
  pack_weight<<<64, 256, 0, stream>>>(Wvs,    pw_proj + 5 * 16384, 120, 112, 128);
  pack_weight<<<64, 256, 0, stream>>>(inv_w1, pw_inv1, 112, 120, 128);
  pack_weight<<<16, 256, 0, stream>>>(rbf_w1, pw_rbf1, 20, 120, 32);
  pack_weight<<<64, 256, 0, stream>>>(inv_w2, pw_inv2, 120, 120, 128);
  pack_weight<<<64, 256, 0, stream>>>(rbf_w2, pw_rbf2, 120, 120, 128);

  node_prep<<<N_NODES / 256, 256, 0, stream>>>(node_scalar, node_equi, ln_w, ln_b, o3w,
                                               ns_pack, nebuf, (float*)d_out);
  proj_gemm<<<(N_NODES / 16) * 6, 256, 0, stream>>>(ns_pack, pw_proj, proj);
  edge_kernel<<<N_EDGES / TE, 256, 0, stream>>>(eidx, rbf, fcut, rsh, nebuf, proj,
                                                pw_inv1, pw_rbf1, pw_inv2, pw_rbf2,
                                                inv_b1, rbf_b1, inv_b2, rbf_b2,
                                                (float*)d_out);
}
